// QKMFA_90400471646423
// MI455X (gfx1250) — compile-verified, matrix operand order checked by reading, not run on hardware
//
#include <hip/hip_runtime.h>
#include <math.h>

#define EPSF 1e-5f
#define BN 16384
#define KC 256
#define DD 128
#define RR 32

typedef _Float16 v16h __attribute__((ext_vector_type(16)));
typedef float    v8f  __attribute__((ext_vector_type(8)));
typedef float    v4f  __attribute__((ext_vector_type(4)));
typedef unsigned int u32;
typedef u32      v4u  __attribute__((ext_vector_type(4)));
typedef u32      v2u  __attribute__((ext_vector_type(2)));

union Frag { v16h h; v4u u[2]; };

__device__ __forceinline__ float softplusf(float x) {
  return x > 20.0f ? x : log1pf(__expf(x));
}

// ---------------------------------------------------------------- psi^-1
__global__ void psi_kernel(const float* __restrict__ psi_rho, float* __restrict__ psh) {
  int d = threadIdx.x;
  if (d < DD) psh[d] = 1.0f / (softplusf(psi_rho[d]) + EPSF);
}

// ---------------------------------------------------------------- X -> f16, TB(b) = -0.5 * sum x^2 psi^-1
__global__ __launch_bounds__(256) void xprep_kernel(const float* __restrict__ x,
    const float* __restrict__ psh, _Float16* __restrict__ X16, float* __restrict__ tb) {
  int wave = threadIdx.x >> 5, lane = threadIdx.x & 31;
  int row = blockIdx.x * 8 + wave;
  int d0 = lane * 4;
  v4f xv = *reinterpret_cast<const v4f*>(x + (size_t)row * DD + d0);
  float t = xv.x * xv.x * psh[d0]     + xv.y * xv.y * psh[d0 + 1]
          + xv.z * xv.z * psh[d0 + 2] + xv.w * xv.w * psh[d0 + 3];
  union { _Float16 h[4]; v2u u; } pk;
  pk.h[0] = (_Float16)xv.x; pk.h[1] = (_Float16)xv.y;
  pk.h[2] = (_Float16)xv.z; pk.h[3] = (_Float16)xv.w;
  *reinterpret_cast<v2u*>(X16 + (size_t)row * DD + d0) = pk.u;
  for (int m = 1; m < 32; m <<= 1) t += __shfl_xor(t, m, 32);
  if (lane == 0) tb[row] = -0.5f * t;
}

// ---------------------------------------------------------------- per-component precompute (1 block per k)
__global__ __launch_bounds__(256) void stageA_kernel(
    const float* __restrict__ mu, const float* __restrict__ dir_raw,
    const float* __restrict__ scale_rho, const float* __restrict__ pi_logits,
    const float* __restrict__ pshg,
    _Float16* __restrict__ Gpack, _Float16* __restrict__ Wmpack,
    float* __restrict__ cbuf, float* __restrict__ CK)
{
  const int k = blockIdx.x;
  const int tid = threadIdx.x;
  __shared__ float psh[DD];
  __shared__ float normp[256];
  __shared__ float scl[RR];
  __shared__ float Ash[DD * 33];
  __shared__ float Msh[RR * 33];
  __shared__ float Zsh[RR * 33];
  __shared__ float Gsh[DD * 33];

  if (tid < DD) psh[tid] = pshg[tid];
  __syncthreads();

  // column norms of dir_raw[k] over D
  {
    int r = tid & 31, g = tid >> 5;
    float s = 0.0f;
    for (int d = g * 16; d < g * 16 + 16; ++d) {
      float v = dir_raw[(size_t)k * DD * RR + (size_t)d * RR + r];
      s += v * v;
    }
    normp[tid] = s;
  }
  __syncthreads();
  if (tid < RR) {
    float s = 0.0f;
    for (int g = 0; g < 8; ++g) s += normp[tid + 32 * g];
    float n = sqrtf(s);
    n = fmaxf(n, EPSF);
    scl[tid] = softplusf(scale_rho[k * RR + tid]) / n;
  }
  __syncthreads();

  // A(d,r) = W(d,r) * sqrt(psi_inv[d])
  for (int e = tid; e < DD * RR; e += 256) {
    int d = e >> 5, r = e & 31;
    float w = dir_raw[(size_t)k * DD * RR + (size_t)d * RR + r] * scl[r];
    Ash[d * 33 + r] = w * sqrtf(psh[d]);
  }
  __syncthreads();

  // M = A^T A + I
  for (int e = tid; e < RR * RR; e += 256) {
    int i = e >> 5, j = e & 31;
    float s = 0.0f;
    for (int d = 0; d < DD; ++d) s += Ash[d * 33 + i] * Ash[d * 33 + j];
    Msh[i * 33 + j] = s + (i == j ? 1.0f : 0.0f);
  }
  __syncthreads();

  // Cholesky (lower, in place)
  for (int j = 0; j < RR; ++j) {
    if (tid == 0) Msh[j * 33 + j] = sqrtf(Msh[j * 33 + j]);
    __syncthreads();
    float dj = Msh[j * 33 + j];
    if (tid > j && tid < RR) Msh[tid * 33 + j] /= dj;
    __syncthreads();
    for (int e = tid; e < RR * RR; e += 256) {
      int i = e >> 5, l = e & 31;
      if (i > j && l > j && l <= i) Msh[i * 33 + l] -= Msh[i * 33 + j] * Msh[l * 33 + j];
    }
    __syncthreads();
  }

  // Z = L^-1 (each thread owns one column, fully independent)
  if (tid < RR) {
    int c = tid;
    Zsh[c * 33 + c] = 1.0f / Msh[c * 33 + c];
    for (int i = c + 1; i < RR; ++i) {
      float s = 0.0f;
      for (int p = c; p < i; ++p) s += Msh[i * 33 + p] * Zsh[p * 33 + c];
      Zsh[i * 33 + c] = -s / Msh[i * 33 + i];
    }
  }
  __syncthreads();

  // G(d,r) = sqrt(psi_inv[d]) * sum_{j<=r} A(d,j) * Z(r,j)
  for (int e = tid; e < DD * RR; e += 256) {
    int d = e >> 5, r = e & 31;
    float s = 0.0f;
    for (int j = 0; j <= r; ++j) s += Ash[d * 33 + j] * Zsh[r * 33 + j];
    Gsh[d * 33 + r] = sqrtf(psh[d]) * s;
  }
  __syncthreads();

  // c_r = sum_d G(d,r) * mu[k,d]
  if (tid < RR) {
    float cc = 0.0f;
    for (int d = 0; d < DD; ++d) cc += Gsh[d * 33 + tid] * mu[k * DD + d];
    cbuf[k * RR + tid] = cc;
  }
  // per-component constant
  if (tid == 0) {
    float ldPsi = 0.0f, ldM = 0.0f, m = 0.0f;
    for (int d = 0; d < DD; ++d) ldPsi -= logf(psh[d]);      // log psi = -log psi_inv
    for (int j = 0; j < RR; ++j)  ldM  += logf(Msh[j * 33 + j]);
    ldM *= 2.0f;
    for (int d = 0; d < DD; ++d) { float mm = mu[k * DD + d]; m += mm * mm * psh[d]; }
    const float C0 = 128.0f * 1.8378770664093453f;           // D*log(2*pi)
    CK[k] = pi_logits[k] - 0.5f * (C0 + ldPsi + ldM + m);
  }

  // Pack G into WMMA B-fragment layout: per k, fragments (t in 0..1, s in 0..3),
  // 32 lanes x 16 f16 each. B element (kk,n): lane L -> n = L&15, kh = L>>4;
  // element i -> kk = 32*s + (i<8 ? 8*kh+i : 16+8*kh+(i-8)); column r = 16*t+n.
  for (int e = tid; e < DD * RR; e += 256) {
    int i  = e & 15;
    int L  = (e >> 4) & 31;
    int s2 = (e >> 9) & 3;
    int t  = (e >> 11) & 1;
    int n  = L & 15, kh = L >> 4;
    int d  = 32 * s2 + (i < 8 ? 8 * kh + i : 16 + 8 * kh + (i - 8));
    int r  = 16 * t + n;
    Gpack[(size_t)k * 4096 + e] = (_Float16)Gsh[d * 33 + r];
  }
  // Pack this component's column of psi^-1*mu into the S-GEMM B fragments
  {
    int kt = k >> 4, n = k & 15;
    for (int d = tid; d < DD; d += 256) {
      int s2 = d >> 5, rem = d & 31, kh, i;
      if (rem < 16) { kh = rem >> 3; i = rem & 7; }
      else          { int r2 = rem - 16; kh = r2 >> 3; i = 8 + (r2 & 7); }
      int L = 16 * kh + n;
      Wmpack[(size_t)(kt * 4 + s2) * 512 + L * 16 + i] = (_Float16)(psh[d] * mu[k * DD + d]);
    }
  }
}

// ---------------------------------------------------------------- fused WMMA GEMM + online logsumexp
// 128 threads = 4 waves; each wave owns TWO 16-row M-subtiles (32 batch rows),
// so every B fragment feeds 2 WMMAs (1 wmma per b128 load).
__global__ __launch_bounds__(128) void fused_kernel(
    const _Float16* __restrict__ X16, const float* __restrict__ tb,
    const _Float16* __restrict__ Gpack, const _Float16* __restrict__ Wmpack,
    const float* __restrict__ cbuf, const float* __restrict__ CK,
    float* __restrict__ lp)
{
  const int wave = threadIdx.x >> 5;
  const int lane = threadIdx.x & 31;
  const int n16  = lane & 15;
  const int kh   = lane >> 4;
  const int rb   = blockIdx.x * 128 + wave * 32;   // 32 rows per wave

  // Two A-fragment sets resident in VGPRs: rows rb..rb+15 and rb+16..rb+31
  Frag a0[4], a1[4];
  {
    const v4u* xr0 = reinterpret_cast<const v4u*>(X16 + (size_t)(rb + n16) * DD);
    const v4u* xr1 = reinterpret_cast<const v4u*>(X16 + (size_t)(rb + 16 + n16) * DD);
#pragma unroll
    for (int s = 0; s < 4; ++s) {
      a0[s].u[0] = xr0[4 * s + kh];
      a0[s].u[1] = xr0[4 * s + kh + 2];
      a1[s].u[0] = xr1[4 * s + kh];
      a1[s].u[1] = xr1[4 * s + kh + 2];
    }
  }

  float rmax0[8], rsum0[8], rmax1[8], rsum1[8];
#pragma unroll
  for (int j = 0; j < 8; ++j) {
    rmax0[j] = -3.0e38f; rsum0[j] = 0.0f;
    rmax1[j] = -3.0e38f; rsum1[j] = 0.0f;
  }

  for (int kt = 0; kt < 16; ++kt) {
    // S tiles: 32 rows x 16 components; each Wm fragment feeds both subtiles
    v8f sacc0 = {0.f,0.f,0.f,0.f,0.f,0.f,0.f,0.f};
    v8f sacc1 = {0.f,0.f,0.f,0.f,0.f,0.f,0.f,0.f};
#pragma unroll
    for (int s = 0; s < 4; ++s) {
      Frag bw;
      const v4u* wp = reinterpret_cast<const v4u*>(Wmpack + ((kt * 4 + s) << 9) + (lane << 4));
      bw.u[0] = wp[0]; bw.u[1] = wp[1];
      sacc0 = __builtin_amdgcn_wmma_f32_16x16x32_f16(false, a0[s].h, false, bw.h,
                                                     (short)0, sacc0, false, false);
      sacc1 = __builtin_amdgcn_wmma_f32_16x16x32_f16(false, a1[s].h, false, bw.h,
                                                     (short)0, sacc1, false, false);
    }
    for (int kk = 0; kk < 16; ++kk) {
      const int k = kt * 16 + kk;
      const v4u* gp = reinterpret_cast<const v4u*>(Gpack + (size_t)k * 4096 + (lane << 4));
      v8f y00 = {0.f,0.f,0.f,0.f,0.f,0.f,0.f,0.f};  // subtile0, r 0..15
      v8f y01 = {0.f,0.f,0.f,0.f,0.f,0.f,0.f,0.f};  // subtile0, r 16..31
      v8f y10 = {0.f,0.f,0.f,0.f,0.f,0.f,0.f,0.f};  // subtile1, r 0..15
      v8f y11 = {0.f,0.f,0.f,0.f,0.f,0.f,0.f,0.f};  // subtile1, r 16..31
#pragma unroll
      for (int s = 0; s < 4; ++s) {
        Frag b0, b1;
        b0.u[0] = gp[s * 64];        b0.u[1] = gp[s * 64 + 1];
        b1.u[0] = gp[256 + s * 64];  b1.u[1] = gp[256 + s * 64 + 1];
        y00 = __builtin_amdgcn_wmma_f32_16x16x32_f16(false, a0[s].h, false, b0.h, (short)0, y00, false, false);
        y10 = __builtin_amdgcn_wmma_f32_16x16x32_f16(false, a1[s].h, false, b0.h, (short)0, y10, false, false);
        y01 = __builtin_amdgcn_wmma_f32_16x16x32_f16(false, a0[s].h, false, b1.h, (short)0, y01, false, false);
        y11 = __builtin_amdgcn_wmma_f32_16x16x32_f16(false, a1[s].h, false, b1.h, (short)0, y11, false, false);
      }
      const float c0 = cbuf[k * RR + n16];
      const float c1 = cbuf[k * RR + 16 + n16];
      float q0[8], q1[8];
#pragma unroll
      for (int j = 0; j < 8; ++j) {
        float u00 = y00[j] - c0, u01 = y01[j] - c1;
        float u10 = y10[j] - c0, u11 = y11[j] - c1;
        q0[j] = u00 * u00 + u01 * u01;
        q1[j] = u10 * u10 + u11 * u11;
      }
      // reduce ||y||^2 over the 16 r-lanes (butterfly within each 16-lane half)
#pragma unroll
      for (int m = 1; m <= 8; m <<= 1) {
#pragma unroll
        for (int j = 0; j < 8; ++j) {
          q0[j] += __shfl_xor(q0[j], m, 32);
          q1[j] += __shfl_xor(q1[j], m, 32);
        }
      }
      const float ckv = CK[k];
      const int src = (lane & 16) | kk;
#pragma unroll
      for (int j = 0; j < 8; ++j) {
        float sv0   = __shfl(sacc0[j], src, 32);
        float sv1   = __shfl(sacc1[j], src, 32);
        float al0   = ckv + sv0 + 0.5f * q0[j];
        float al1   = ckv + sv1 + 0.5f * q1[j];
        float nm0   = fmaxf(rmax0[j], al0);
        float nm1   = fmaxf(rmax1[j], al1);
        rsum0[j]    = rsum0[j] * __expf(rmax0[j] - nm0) + __expf(al0 - nm0);
        rsum1[j]    = rsum1[j] * __expf(rmax1[j] - nm1) + __expf(al1 - nm1);
        rmax0[j]    = nm0;
        rmax1[j]    = nm1;
      }
    }
  }
  if ((lane & 15) == 0) {   // lane 0 -> rows 0..7, lane 16 -> rows 8..15 (per subtile)
    const int rA = rb + (kh << 3);
    const int rB = rb + 16 + (kh << 3);
#pragma unroll
    for (int j = 0; j < 8; ++j) {
      lp[rA + j] = tb[rA + j] + rmax0[j] + __logf(rsum0[j]);
      lp[rB + j] = tb[rB + j] + rmax1[j] + __logf(rsum1[j]);
    }
  }
}

// ---------------------------------------------------------------- final scalar
__global__ __launch_bounds__(256) void final_kernel(const float* __restrict__ lp,
    const float* __restrict__ pi_logits, float* __restrict__ out) {
  __shared__ float sm[256];
  int tid = threadIdx.x;
  float v = pi_logits[tid];
  sm[tid] = v; __syncthreads();
  for (int s = 128; s > 0; s >>= 1) { if (tid < s) sm[tid] = fmaxf(sm[tid], sm[tid + s]); __syncthreads(); }
  float mx = sm[0]; __syncthreads();
  sm[tid] = __expf(v - mx); __syncthreads();
  for (int s = 128; s > 0; s >>= 1) { if (tid < s) sm[tid] += sm[tid + s]; __syncthreads(); }
  float logZ = mx + __logf(sm[0]); __syncthreads();
  float acc = 0.0f;
  for (int i = tid; i < BN; i += 256) acc += lp[i];
  sm[tid] = acc; __syncthreads();
  for (int s = 128; s > 0; s >>= 1) { if (tid < s) sm[tid] += sm[tid + s]; __syncthreads(); }
  if (tid == 0) out[0] = -(sm[0] / (float)BN) + logZ;
}

// ---------------------------------------------------------------- launcher
extern "C" void kernel_launch(void* const* d_in, const int* in_sizes, int n_in,
                              void* d_out, int out_size, void* d_ws, size_t ws_size,
                              hipStream_t stream) {
  (void)in_sizes; (void)n_in; (void)out_size; (void)ws_size;
  const float* x         = (const float*)d_in[0];
  const float* mu        = (const float*)d_in[1];
  const float* dir_raw   = (const float*)d_in[2];
  const float* scale_rho = (const float*)d_in[3];
  const float* psi_rho   = (const float*)d_in[4];
  const float* pi_logits = (const float*)d_in[5];
  float* out = (float*)d_out;

  char* w = (char*)d_ws;
  float*    psh    = (float*)w;      w += 512;                       // 128 f32
  float*    tb     = (float*)w;      w += (size_t)BN * 4;            // 64 KB
  float*    lp     = (float*)w;      w += (size_t)BN * 4;            // 64 KB
  float*    cbuf   = (float*)w;      w += (size_t)KC * RR * 4;       // 32 KB
  float*    CK     = (float*)w;      w += (size_t)KC * 4;            // 1 KB
  _Float16* X16    = (_Float16*)w;   w += (size_t)BN * DD * 2;       // 4 MB
  _Float16* Gpack  = (_Float16*)w;   w += (size_t)KC * 4096 * 2;     // 2 MB
  _Float16* Wmpack = (_Float16*)w;   w += (size_t)16 * 4 * 512 * 2;  // 64 KB

  psi_kernel<<<1, 128, 0, stream>>>(psi_rho, psh);
  xprep_kernel<<<BN / 8, 256, 0, stream>>>(x, psh, X16, tb);
  stageA_kernel<<<KC, 256, 0, stream>>>(mu, dir_raw, scale_rho, pi_logits, psh,
                                        Gpack, Wmpack, cbuf, CK);
  fused_kernel<<<BN / 128, 128, 0, stream>>>(X16, tb, Gpack, Wmpack, cbuf, CK, lp);
  final_kernel<<<1, 256, 0, stream>>>(lp, pi_logits, out);
}
